// GumbelVQ_49804440764750
// MI455X (gfx1250) — matrix operand order repaired and checked by reading, not compiled, not run on hardware
//
#include <hip/hip_runtime.h>
#include <hip/hip_bf16.h>

typedef __attribute__((ext_vector_type(16))) __bf16 v16bf;
typedef __attribute__((ext_vector_type(8)))  float  v8f;

#define K_CB   4096
#define D_DIM  256
#define N_TOK  32768
#define ROWS   16
#define LSTR   4100   // padded row stride (floats) for the logits LDS array

// ---------------------------------------------------------------------------
// Prep: codebook f32 (K x D) -> bf16 row-major (GEMM1 B), bf16 transposed
// (GEMM2 B), and per-code squared norms. One-time ~4MB, stays L2-resident.
// ---------------------------------------------------------------------------
__global__ __launch_bounds__(256) void vq_prep(const float* __restrict__ cb,
                                               __bf16* __restrict__ cbb,
                                               __bf16* __restrict__ cbt,
                                               float* __restrict__ cnorm) {
  const int k = blockIdx.x;
  const int d = threadIdx.x;
  const float v = cb[(size_t)k * D_DIM + d];
  cbb[(size_t)k * D_DIM + d] = (__bf16)v;
  cbt[(size_t)d * K_CB + k]  = (__bf16)v;
  __shared__ float red[256];
  red[d] = v * v;
  __syncthreads();
  for (int s = 128; s > 0; s >>= 1) {
    if (d < s) red[d] += red[d + s];
    __syncthreads();
  }
  if (d == 0) cnorm[k] = red[0];
}

// ---------------------------------------------------------------------------
// Fused main kernel: one workgroup (8 wave32) per 16-token tile.
//  GEMM1 (bf16 WMMA) -> dist/argmin/logits -> softmax -> encodings ->
//  GEMM2 (bf16 WMMA) -> quantized.
// ---------------------------------------------------------------------------
__global__ __launch_bounds__(256) void vq_main(const float* __restrict__ x,
                                               const float* __restrict__ noise,
                                               const __bf16* __restrict__ cbb,
                                               const __bf16* __restrict__ cbt,
                                               const float* __restrict__ cnorm,
                                               float* __restrict__ outQ,
                                               float* __restrict__ outE,
                                               float* __restrict__ outI) {
  __shared__ float sLog[ROWS * LSTR];                 // 256 KB logits / probs
  __shared__ __align__(32) __bf16 sXb[ROWS * D_DIM];  // x tile, WMMA-A swizzled
  __shared__ float sXn[ROWS];                         // |x|^2 per row
  __shared__ float sCandV[ROWS * 128];                // argmin candidates
  __shared__ int   sCandI[ROWS * 128];
  __shared__ float sRed[ROWS * 16];                   // reduction scratch
  __shared__ float sRowMax[ROWS];
  __shared__ float sRowInv[ROWS];

  const int tid  = threadIdx.x;
  const int wave = tid >> 5;
  const int lane = tid & 31;
  const int nn   = lane & 15;   // N-lane / M-lane index
  const int hf   = lane >> 4;   // which K/M half this lane holds
  const int rowbase = blockIdx.x * ROWS;

  if (tid < ROWS) sXn[tid] = 0.0f;
  __syncthreads();

  // ---- Phase 0: stage x tile (coalesced), bf16 + A-layout swizzle, row norms
  for (int i = 0; i < ROWS; ++i) {
    const float v = x[(size_t)(rowbase + i) * D_DIM + tid];
    const int c  = tid >> 5;          // 32-wide K chunk
    const int dd = tid & 31;
    const int h  = dd >> 4;           // lane half that owns this K
    const int j  = dd & 15;           // slot within lane's v16bf
    sXb[i * D_DIM + c * 32 + h * 16 + j] = (__bf16)v;
    atomicAdd(&sXn[i], v * v);        // ds_add_f32
  }
  __syncthreads();

  // A fragments (x tile) resident in registers for all 32 GEMM1 iterations
  v16bf afrag[8];
#pragma unroll
  for (int c = 0; c < 8; ++c)
    afrag[c] = *(const v16bf*)&sXb[nn * D_DIM + c * 32 + hf * 16];

  float xnr[8];
#pragma unroll
  for (int r = 0; r < 8; ++r) xnr[r] = sXn[r + hf * 8];

  const float* nrow[8];
#pragma unroll
  for (int r = 0; r < 8; ++r)
    nrow[r] = noise + (size_t)(rowbase + r + hf * 8) * K_CB;

  float bestV[8];
  int   bestI[8];
#pragma unroll
  for (int r = 0; r < 8; ++r) { bestV[r] = -3.4e38f; bestI[r] = 0; }

  // ---- Phase 1: GEMM1 over all 4096 codes; logits + argmin tracking
  for (int it = 0; it < 32; ++it) {
    const int kb  = it * 128 + wave * 16;
    const int col = kb + nn;
    v8f acc = {};
#pragma unroll
    for (int c = 0; c < 8; ++c) {
      v16bf b = *(const v16bf*)&cbb[(size_t)col * D_DIM + c * 32 + hf * 16];
      acc = __builtin_amdgcn_wmma_f32_16x16x32_bf16(false, afrag[c], false, b,
                                                    (short)0, acc, false, false);
    }
    const float cn = cnorm[col];
#pragma unroll
    for (int r = 0; r < 8; ++r) {
      const float dot = acc[r];
      const float sv  = dot - 0.5f * cn;        // argmin(dist) == argmax(sv)
      if (sv > bestV[r]) { bestV[r] = sv; bestI[r] = col; }
      const float sq   = xnr[r] + cn - 2.0f * dot;
      const float dist = sqrtf(fmaxf(sq, 0.0f));
      sLog[(r + hf * 8) * LSTR + col] = nrow[r][col] - dist;  // TAU == 1
    }
  }
#pragma unroll
  for (int r = 0; r < 8; ++r) {
    sCandV[(r + hf * 8) * 128 + wave * 16 + nn] = bestV[r];
    sCandI[(r + hf * 8) * 128 + wave * 16 + nn] = bestI[r];
  }
  __syncthreads();

  // ---- Phase 2: row max, argmin finalize, sum-exp, normalize, store encodings
  const int row = tid >> 4;
  const int j0  = tid & 15;
  float mx = -3.4e38f;
  for (int i = 0; i < 256; ++i)
    mx = fmaxf(mx, sLog[row * LSTR + j0 + 16 * i]);
  sRed[row * 16 + j0] = mx;
  __syncthreads();
  if (tid < ROWS) {
    float m = -3.4e38f;
    for (int j = 0; j < 16; ++j) m = fmaxf(m, sRed[tid * 16 + j]);
    sRowMax[tid] = m;
    float bv = -3.4e38f; int bi = 0x7fffffff;
    for (int j = 0; j < 128; ++j) {
      const float v  = sCandV[tid * 128 + j];
      const int   ix = sCandI[tid * 128 + j];
      if (v > bv || (v == bv && ix < bi)) { bv = v; bi = ix; }
    }
    outI[rowbase + tid] = (float)bi;
  }
  __syncthreads();
  const float rm = sRowMax[row];
  float s = 0.0f;
  for (int i = 0; i < 256; ++i) {
    float* p = &sLog[row * LSTR + j0 + 16 * i];
    const float e = __expf(*p - rm);
    *p = e;
    s += e;
  }
  sRed[row * 16 + j0] = s;
  __syncthreads();
  if (tid < ROWS) {
    float t = 0.0f;
    for (int j = 0; j < 16; ++j) t += sRed[tid * 16 + j];
    sRowInv[tid] = 1.0f / t;
  }
  __syncthreads();
  for (int i = 0; i < 256; ++i) {           // coalesced encodings store
    const int idx = tid + i * 256;
    const int r2  = idx >> 12;
    const int c2  = idx & 4095;
    const float p = sLog[r2 * LSTR + c2] * sRowInv[r2];
    sLog[r2 * LSTR + c2] = p;
    outE[(size_t)(rowbase + r2) * K_CB + c2] = p;
  }
  __syncthreads();

  // ---- Phase 3: quantized = P (16x4096) @ codebook (4096x256), bf16 WMMA.
  // Each wave owns two 16-wide D tiles.
  v8f acc0 = {}, acc1 = {};
  const int d0 = wave * 32;
  for (int kc = 0; kc < 128; ++kc) {
    const float* s0 = &sLog[nn * LSTR + kc * 32 + hf * 16];
    v16bf a2;
#pragma unroll
    for (int j = 0; j < 8; ++j) {
      a2[j]     = (__bf16)s0[j];        // K = 32*kc + 16*hf + j
      a2[j + 8] = (__bf16)s0[j + 8];    // K = 32*kc + 16*hf + 8 + j
    }
    v16bf b0 = *(const v16bf*)&cbt[(size_t)(d0 + nn) * K_CB + kc * 32 + hf * 16];
    v16bf b1 = *(const v16bf*)&cbt[(size_t)(d0 + 16 + nn) * K_CB + kc * 32 + hf * 16];
    acc0 = __builtin_amdgcn_wmma_f32_16x16x32_bf16(false, a2, false, b0,
                                                   (short)0, acc0, false, false);
    acc1 = __builtin_amdgcn_wmma_f32_16x16x32_bf16(false, a2, false, b1,
                                                   (short)0, acc1, false, false);
  }
#pragma unroll
  for (int r = 0; r < 8; ++r) {
    const int mm = r + hf * 8;
    outQ[(size_t)(rowbase + mm) * D_DIM + d0 + nn]      = acc0[r];
    outQ[(size_t)(rowbase + mm) * D_DIM + d0 + 16 + nn] = acc1[r];
  }
}

// ---------------------------------------------------------------------------
extern "C" void kernel_launch(void* const* d_in, const int* in_sizes, int n_in,
                              void* d_out, int out_size, void* d_ws, size_t ws_size,
                              hipStream_t stream) {
  (void)in_sizes; (void)n_in; (void)out_size; (void)ws_size;
  const float* x     = (const float*)d_in[0];
  const float* cb    = (const float*)d_in[1];
  const float* noise = (const float*)d_in[2];

  __bf16* cbb = (__bf16*)d_ws;                        // K*D bf16 (2 MB)
  __bf16* cbt = cbb + (size_t)K_CB * D_DIM;           // D*K bf16 (2 MB)
  float*  cn  = (float*)(cbt + (size_t)D_DIM * K_CB); // K floats

  float* outQ = (float*)d_out;                        // quantized (N*D)
  float* outE = outQ + (size_t)N_TOK * D_DIM;         // encodings (N*K)
  float* outI = outE + (size_t)N_TOK * K_CB;          // indices   (N)

  vq_prep<<<K_CB, 256, 0, stream>>>(cb, cbb, cbt, cn);
  vq_main<<<N_TOK / ROWS, 256, 0, stream>>>(x, noise, cbb, cbt, cn,
                                            outQ, outE, outI);
}
